// MambaFullCALayer_42812234006553
// MI455X (gfx1250) — compile-verified, matrix operand order checked by reading, not run on hardware
//
#include <hip/hip_runtime.h>
#include <hip/hip_bf16.h>
#include <math.h>

// ---------------- model constants (match reference) ----------------
#define DMODEL   512
#define DINNER   1024
#define DSTATE   128
#define DTRANK   32
#define DCONV    4
#define NHEADS   8
#define HDIM     64
#define NKV      8064      // sum of SPATIAL level sizes
#define BSZ      2
#define SLEN     256
#define ROWS_KV  (BSZ * NKV)        // 16128
#define ROWS_OUT (BSZ * SLEN)       // 512

// block-tile geometry for the KV GEMM
#define BLK_M 128
#define BLK_N 64
#define BLK_K 32
#define KCHUNKS (DMODEL / BLK_K)    // 16

typedef _Float16 half8  __attribute__((ext_vector_type(8)));
typedef _Float16 half16 __attribute__((ext_vector_type(16)));
typedef float    float8 __attribute__((ext_vector_type(8)));
typedef unsigned int uint4v __attribute__((ext_vector_type(4)));
typedef int      int4v  __attribute__((ext_vector_type(4)));
typedef int      int8v  __attribute__((ext_vector_type(8)));

#if __has_builtin(__builtin_amdgcn_tensor_load_to_lds) && __has_builtin(__builtin_amdgcn_s_wait_tensorcnt)
#define USE_TDM 1
#else
#define USE_TDM 0
#endif

// ---------------- generic f32 -> f16 cast ----------------
__global__ void cast_f32_f16(const float* __restrict__ in, _Float16* __restrict__ out, int n) {
    int i = blockIdx.x * blockDim.x + threadIdx.x;
    if (i < n) out[i] = (_Float16)in[i];
}

// ---------------- A_neg = -exp(A_log) ----------------
__global__ void aneg_kernel(const float* __restrict__ A_log, float* __restrict__ Aneg, int n) {
    int i = blockIdx.x * blockDim.x + threadIdx.x;
    if (i < n) Aneg[i] = -__expf(A_log[i]);
}

// ---------------- time positional encoding ----------------
// SPATIAL = [(8,512),(8,256),(8,128),(8,64),(8,32),(8,16)] -> level starts/widths hardcoded
__device__ __forceinline__ float time_pe_val(int n, int c512) {
    int start, W;
    if      (n < 4096) { start = 0;    W = 512; }
    else if (n < 6144) { start = 4096; W = 256; }
    else if (n < 7168) { start = 6144; W = 128; }
    else if (n < 7680) { start = 7168; W = 64;  }
    else if (n < 7936) { start = 7680; W = 32;  }
    else               { start = 7936; W = 16;  }
    float tn  = (float)((n - start) % W) / (float)(W - 1);
    float pos = tn * (float)NKV;
    int idx   = c512 & 255;
    float freq = __expf(-(float)idx * (9.210340371976184f / 256.0f)); // ln(10000)/256
    float ang  = pos * freq;
    return (c512 < 256) ? sinf(ang) : cosf(ang);
}

#if USE_TDM
// Issue a 2D TDM tile load: rows x cols f16 elements, row stride strideElems,
// from global base gbase into LDS byte address ldsAddr. Wave-level op.
__device__ __forceinline__ void tdm_load_2d(const _Float16* gbase, unsigned ldsAddr,
                                            int rows, int cols, int strideElems) {
    unsigned long long ga = (unsigned long long)(size_t)gbase;
    uint4v g0;
    g0[0] = 1u;                                   // count=1, user mode, no gather
    g0[1] = ldsAddr;                              // lds_addr (bytes)
    g0[2] = (unsigned)(ga & 0xFFFFFFFFu);         // global_addr[31:0]
    g0[3] = (unsigned)((ga >> 32) & 0x01FFFFFFu)  // global_addr[56:32]
          | (2u << 30);                           // type = 2 ("image")
    unsigned td0 = (unsigned)strideElems;         // tensor_dim0 (OOB bound)
    unsigned td1 = 0x7FFFFFu;                     // tensor_dim1 (large, never OOB)
    int8v g1;
    g1[0] = (int)(1u << 16);                      // wg_mask=0, data_size=1 (2 bytes)
    g1[1] = (int)((td0 & 0xFFFFu) << 16);         // tensor_dim0[15:0] @ bits 63:48
    g1[2] = (int)((td0 >> 16) | ((td1 & 0xFFFFu) << 16));   // td0[31:16] | td1[15:0]
    g1[3] = (int)((td1 >> 16) | ((unsigned)cols << 16));    // td1[31:16] | tile_dim0
    g1[4] = (int)((unsigned)rows);                // tile_dim1 | tile_dim2=0
    g1[5] = (int)((unsigned)strideElems);         // tensor_dim0_stride[31:0]
    g1[6] = 0;                                    // stride0 hi | stride1 lo
    g1[7] = 0;                                    // stride1 hi
    int4v gz = {0, 0, 0, 0};
#if __clang_major__ >= 23
    int8v gz8 = {0, 0, 0, 0, 0, 0, 0, 0};
    __builtin_amdgcn_tensor_load_to_lds(g0, g1, gz, gz, gz8, 0);
#else
    __builtin_amdgcn_tensor_load_to_lds(g0, g1, gz, gz, 0);
#endif
}
#endif

// ---------------- KV projection GEMM (LDS-tiled WMMA, TDM staging) ----------------
// C[row,col] = sum_k memory_f16[row,k] * kv_w_f16[col,k]; K = 512
// block tile 128M x 64N, 8 waves, each wave a 32x32 micro-tile (2x2 wmma frags)
// epilogue: + kv_b[col]; K-half gets time_pe; scatter into K/V f16 [b,h,n,hd]
__global__ void __launch_bounds__(256)
kv_gemm_wmma(const _Float16* __restrict__ A,   // [16128,512]
             const _Float16* __restrict__ W,   // [1024,512]
             const float*    __restrict__ kv_b,
             _Float16* __restrict__ Kf,        // [B,H,NKV,HDIM]
             _Float16* __restrict__ Vf) {
    __shared__ _Float16 sA[2][BLK_M * BLK_K];  // 2 x 8 KB
    __shared__ _Float16 sB[2][BLK_N * BLK_K];  // 2 x 4 KB

    const int tid   = threadIdx.x;
    const int wid   = tid >> 5;
    const int lane  = tid & 31;
    const int blkM  = blockIdx.x >> 4;          // 0..125  (16128/128)
    const int blkN  = blockIdx.x & 15;          // 0..15   (1024/64)
    const int waveM = wid >> 1;                 // 0..3 -> 32-row strip
    const int waveN = wid & 1;                  // 0..1 -> 32-col strip

    const _Float16* aTile = A + (size_t)blkM * BLK_M * DMODEL;  // +chunk*32
    const _Float16* bTile = W + (size_t)blkN * BLK_N * DMODEL;

    float8 acc[2][2] = {};

    // ---- stage chunk 0 into buffer 0 ----
#if USE_TDM
    if (wid == 0) {
        tdm_load_2d(aTile, (unsigned)(size_t)&sA[0][0], BLK_M, BLK_K, DMODEL);
        tdm_load_2d(bTile, (unsigned)(size_t)&sB[0][0], BLK_N, BLK_K, DMODEL);
        __builtin_amdgcn_s_wait_tensorcnt(0);
    }
#else
    {   // cooperative copy: A 4096 halves (2 x half8/thread), B 2048 halves (1 x half8/thread)
        int e = tid * 8;
        *(half8*)&sA[0][e]        = *(const half8*)(aTile + (e >> 5) * DMODEL + (e & 31));
        int e2 = e + 2048;
        *(half8*)&sA[0][e2]       = *(const half8*)(aTile + (e2 >> 5) * DMODEL + (e2 & 31));
        *(half8*)&sB[0][e]        = *(const half8*)(bTile + (e >> 5) * DMODEL + (e & 31));
    }
#endif
    __syncthreads();

    const int r    = lane & 15;
    const int koff = (lane & 16) ? 8 : 0;

    for (int ch = 0; ch < KCHUNKS; ++ch) {
        int buf = ch & 1;
        // ---- prefetch next chunk into the other buffer ----
        if (ch + 1 < KCHUNKS) {
            int nk = (ch + 1) * BLK_K;
#if USE_TDM
            if (wid == 0) {
                tdm_load_2d(aTile + nk, (unsigned)(size_t)&sA[buf ^ 1][0], BLK_M, BLK_K, DMODEL);
                tdm_load_2d(bTile + nk, (unsigned)(size_t)&sB[buf ^ 1][0], BLK_N, BLK_K, DMODEL);
            }
#else
            int e = tid * 8;
            *(half8*)&sA[buf ^ 1][e]  = *(const half8*)(aTile + (e >> 5) * DMODEL + nk + (e & 31));
            int e2 = e + 2048;
            *(half8*)&sA[buf ^ 1][e2] = *(const half8*)(aTile + (e2 >> 5) * DMODEL + nk + (e2 & 31));
            *(half8*)&sB[buf ^ 1][e]  = *(const half8*)(bTile + (e >> 5) * DMODEL + nk + (e & 31));
#endif
        }
        // ---- build fragments from LDS, 4 WMMAs ----
        half16 af[2], bf[2];
        #pragma unroll
        for (int mi = 0; mi < 2; ++mi) {
            const _Float16* p = &sA[buf][(waveM * 32 + mi * 16 + r) * BLK_K + koff];
            half8 lo = *(const half8*)p;
            half8 hi = *(const half8*)(p + 16);
            #pragma unroll
            for (int i = 0; i < 8; ++i) { af[mi][i] = lo[i]; af[mi][8 + i] = hi[i]; }
        }
        #pragma unroll
        for (int ni = 0; ni < 2; ++ni) {
            const _Float16* p = &sB[buf][(waveN * 32 + ni * 16 + r) * BLK_K + koff];
            half8 lo = *(const half8*)p;
            half8 hi = *(const half8*)(p + 16);
            #pragma unroll
            for (int i = 0; i < 8; ++i) { bf[ni][i] = lo[i]; bf[ni][8 + i] = hi[i]; }
        }
        #pragma unroll
        for (int mi = 0; mi < 2; ++mi)
            #pragma unroll
            for (int ni = 0; ni < 2; ++ni)
                acc[mi][ni] = __builtin_amdgcn_wmma_f32_16x16x32_f16(
                    false, af[mi], false, bf[ni], (short)0, acc[mi][ni], false, false);

#if USE_TDM
        if (wid == 0 && ch + 1 < KCHUNKS) __builtin_amdgcn_s_wait_tensorcnt(0);
#endif
        __syncthreads();
    }

    // ---- epilogue: bias + PE, scatter to K/V ----
    const int rowHi = (lane & 16) ? 8 : 0;
    #pragma unroll
    for (int mi = 0; mi < 2; ++mi) {
        #pragma unroll
        for (int ni = 0; ni < 2; ++ni) {
            int col   = blkN * BLK_N + waveN * 32 + ni * 16 + (lane & 15);
            int c512  = col & 511;
            int kvsel = col >> 9;    // 0 = K, 1 = V
            int h     = c512 >> 6;
            int hd    = c512 & 63;
            float bias = kv_b[col];
            #pragma unroll
            for (int rr = 0; rr < 8; ++rr) {
                int row = blkM * BLK_M + waveM * 32 + mi * 16 + rr + rowHi;
                int b   = row / NKV;
                int n   = row % NKV;
                float val = acc[mi][ni][rr] + bias;
                size_t dst = ((size_t)(b * NHEADS + h) * NKV + n) * HDIM + hd;
                if (kvsel == 0) {
                    val += time_pe_val(n, c512);
                    Kf[dst] = (_Float16)val;
                } else {
                    Vf[dst] = (_Float16)val;
                }
            }
        }
    }
}

// ---------------- output projection GEMM (WMMA f16) ----------------
// out[row,col] = sum_k fused_f16[row,k]*outp_w_f16[col,k] + outp_b[col]; K=1024
__global__ void out_gemm_wmma(const _Float16* __restrict__ A,   // [512,1024]
                              const _Float16* __restrict__ W,   // [512,1024]
                              const float*    __restrict__ bias,
                              float* __restrict__ out) {
    int wid  = blockIdx.x * (blockDim.x >> 5) + (threadIdx.x >> 5);
    int lane = threadIdx.x & 31;
    int tileM = wid >> 5;   // 0..31
    int tileN = wid & 31;   // 0..31
    if (tileM >= 32) return;

    int r    = lane & 15;
    int koff = (lane & 16) ? 8 : 0;
    const _Float16* aBase = A + (size_t)(tileM * 16 + r) * 1024 + koff;
    const _Float16* bBase = W + (size_t)(tileN * 16 + r) * 1024 + koff;

    float8 c = {};
    #pragma unroll 4
    for (int kk = 0; kk < 1024; kk += 32) {
        half8 a0 = *(const half8*)(aBase + kk);
        half8 a1 = *(const half8*)(aBase + kk + 16);
        half8 b0 = *(const half8*)(bBase + kk);
        half8 b1 = *(const half8*)(bBase + kk + 16);
        half16 a = {}, b = {};
        #pragma unroll
        for (int i = 0; i < 8; ++i) { a[i] = a0[i]; a[8 + i] = a1[i]; b[i] = b0[i]; b[8 + i] = b1[i]; }
        c = __builtin_amdgcn_wmma_f32_16x16x32_f16(false, a, false, b, (short)0, c, false, false);
    }

    int rowHi = (lane & 16) ? 8 : 0;
    int col   = tileN * 16 + (lane & 15);
    #pragma unroll
    for (int rr = 0; rr < 8; ++rr) {
        int row = tileM * 16 + rr + rowHi;
        out[(size_t)row * DMODEL + col] = c[rr] + bias[col];
    }
}

// ---------------- block reductions (1024 threads) ----------------
__device__ __forceinline__ float blockSum(float v, float* s) {
    int tid = threadIdx.x;
    s[tid] = v; __syncthreads();
    for (int o = 512; o > 0; o >>= 1) {
        if (tid < o) s[tid] += s[tid + o];
        __syncthreads();
    }
    float r = s[0]; __syncthreads();
    return r;
}
__device__ __forceinline__ float blockMax(float v, float* s) {
    int tid = threadIdx.x;
    s[tid] = v; __syncthreads();
    for (int o = 512; o > 0; o >>= 1) {
        if (tid < o) s[tid] = fmaxf(s[tid], s[tid + o]);
        __syncthreads();
    }
    float r = s[0]; __syncthreads();
    return r;
}

__device__ __forceinline__ float softplus_f(float x) {
    return (x > 20.0f) ? x : log1pf(__expf(x));
}
__device__ __forceinline__ float silu_f(float x) {
    return x / (1.0f + __expf(-x));
}

struct MambaBuf { float x[DINNER]; float z[DINNER]; float xdb[320]; };
union  ScanU    { float logits[NKV]; MambaBuf m; };

// ---------------- persistent per-batch sequential scan ----------------
// one block per batch element; 1024 threads = 32 waves on one WGP
__global__ void __launch_bounds__(1024)
scan_kernel(const float* __restrict__ tgt,          // [B,S,512]
            const float* __restrict__ enc,          // [B,1,512]
            const float* __restrict__ fusion_w, const float* __restrict__ fusion_b,
            const float* __restrict__ n1g, const float* __restrict__ n1b,
            const float* __restrict__ n2g, const float* __restrict__ n2b,
            const float* __restrict__ q_w, const float* __restrict__ q_b,
            const float* __restrict__ cao_w, const float* __restrict__ cao_b,
            const float* __restrict__ in_proj_w,    // [2048,512]
            const float* __restrict__ conv_w,       // [1024,4]
            const float* __restrict__ conv_b,
            const float* __restrict__ x_proj_w,     // [288,1024]
            const float* __restrict__ dt_w,         // [1024,32]
            const float* __restrict__ dt_b,
            const float* __restrict__ Aneg,         // [1024,128]
            const float* __restrict__ Dp,
            const float* __restrict__ m_out_w,      // [512,1024]
            const _Float16* __restrict__ Kf,        // [B,H,NKV,HDIM]
            const _Float16* __restrict__ Vf,
            float* __restrict__ ssm_ws,             // [B,128,1024] layout [s][d]
            _Float16* __restrict__ fused_f16)       // [B,S,1024]
{
    const int b   = blockIdx.x;
    const int tid = threadIdx.x;

    __shared__ float s_prev[DMODEL];
    __shared__ float s_ln[DMODEL];     // LN2 output, reused as attention output o
    __shared__ float s_q[DMODEL];      // q (pre-scaled), reused as ca
    __shared__ float s_proj[DMODEL];
    __shared__ float s_min[DMODEL];    // mamba input (LN1 of proj)
    __shared__ float s_red[1024];
    __shared__ float s_conv[DINNER * DCONV];
    __shared__ ScanU s_u;

    float* ssm = ssm_ws + (size_t)b * DSTATE * DINNER;

    // deterministic init each launch
    for (int i = tid; i < DSTATE * DINNER; i += 1024) ssm[i] = 0.0f;
    for (int i = tid; i < DINNER * DCONV; i += 1024) s_conv[i] = 0.0f;
    if (tid < DMODEL) s_prev[tid] = enc[b * DMODEL + tid];
    __syncthreads();

    const float qscale = 0.125f; // 1/sqrt(64)

    for (int t = 0; t < SLEN; ++t) {
        // ---- LN2(prev) ----
        float xv = (tid < DMODEL) ? s_prev[tid] : 0.0f;
        float mu = blockSum(xv, s_red) * (1.0f / DMODEL);
        float dv = (tid < DMODEL) ? (s_prev[tid] - mu) : 0.0f;
        float var = blockSum(dv * dv, s_red) * (1.0f / DMODEL);
        float rstd = rsqrtf(var + 1e-5f);
        if (tid < DMODEL) s_ln[tid] = (s_prev[tid] - mu) * rstd * n2g[tid] + n2b[tid];
        __syncthreads();

        // ---- q = LN2(prev) @ q_w.T + q_b  (scale folded in) ----
        if (tid < DMODEL) {
            const float* w = q_w + (size_t)tid * DMODEL;
            __builtin_prefetch(w, 0, 3);
            float acc = q_b[tid];
            #pragma unroll 8
            for (int k = 0; k < DMODEL; ++k) acc += s_ln[k] * w[k];
            s_q[tid] = acc * qscale;
        }
        __syncthreads();

        // ---- per-head attention over NKV keys ----
        for (int h = 0; h < NHEADS; ++h) {
            const _Float16* kb = Kf + ((size_t)(b * NHEADS + h) * NKV) * HDIM;
            float lmax = -1e30f;
            for (int n = tid; n < NKV; n += 1024) {
                const _Float16* kp = kb + (size_t)n * HDIM;
                float acc = 0.0f;
                #pragma unroll 8
                for (int d = 0; d < HDIM; ++d) acc += s_q[h * HDIM + d] * (float)kp[d];
                s_u.logits[n] = acc;
                lmax = fmaxf(lmax, acc);
            }
            float M = blockMax(lmax, s_red);
            float lsum = 0.0f;
            for (int n = tid; n < NKV; n += 1024) {
                float p = __expf(s_u.logits[n] - M);
                s_u.logits[n] = p;
                lsum += p;
            }
            float L = blockSum(lsum, s_red);
            float invL = 1.0f / L;
            // p @ v : thread (j,hd), j = tid>>6 strides n, hd = tid&63
            int j = tid >> 6, hd = tid & 63;
            const _Float16* vp = Vf + ((size_t)(b * NHEADS + h) * NKV) * HDIM + hd;
            float acc = 0.0f;
            for (int n = j; n < NKV; n += 16) acc += s_u.logits[n] * (float)vp[(size_t)n * HDIM];
            s_red[tid] = acc;
            __syncthreads();
            if (tid < HDIM) {
                float o = 0.0f;
                #pragma unroll
                for (int jj = 0; jj < 16; ++jj) o += s_red[jj * 64 + tid];
                s_ln[h * HDIM + tid] = o * invL;
            }
            __syncthreads();
        }

        // ---- ca = o @ cao_w.T + cao_b  (into s_q; also write fused[:,512:]) ----
        if (tid < DMODEL) {
            const float* w = cao_w + (size_t)tid * DMODEL;
            float acc = cao_b[tid];
            #pragma unroll 8
            for (int k = 0; k < DMODEL; ++k) acc += s_ln[k] * w[k];
            s_q[tid] = acc;
            fused_f16[((size_t)(b * SLEN + t)) * 1024 + DMODEL + tid] = (_Float16)acc;
        }
        __syncthreads();

        // ---- proj = [token_t, ca] @ fusion_w.T + fusion_b ----
        const float* tok = tgt + ((size_t)(b * SLEN + t)) * DMODEL;
        if (tid < DMODEL) {
            const float* w = fusion_w + (size_t)tid * (2 * DMODEL);
            float acc = fusion_b[tid];
            #pragma unroll 8
            for (int k = 0; k < DMODEL; ++k) acc += tok[k] * w[k];
            #pragma unroll 8
            for (int k = 0; k < DMODEL; ++k) acc += s_q[k] * w[DMODEL + k];
            s_proj[tid] = acc;
        }
        __syncthreads();

        // ---- LN1(proj) -> m_in ----
        xv = (tid < DMODEL) ? s_proj[tid] : 0.0f;
        mu = blockSum(xv, s_red) * (1.0f / DMODEL);
        dv = (tid < DMODEL) ? (s_proj[tid] - mu) : 0.0f;
        var = blockSum(dv * dv, s_red) * (1.0f / DMODEL);
        rstd = rsqrtf(var + 1e-5f);
        if (tid < DMODEL) s_min[tid] = (s_proj[tid] - mu) * rstd * n1g[tid] + n1b[tid];
        __syncthreads();

        // ---- in_proj: 2048 outputs, 2 per thread ----
        {
            const float* w0 = in_proj_w + (size_t)tid * DMODEL;
            const float* w1 = in_proj_w + (size_t)(tid + DINNER) * DMODEL;
            __builtin_prefetch(w0, 0, 3);
            float a0 = 0.0f, a1 = 0.0f;
            #pragma unroll 8
            for (int k = 0; k < DMODEL; ++k) { float m = s_min[k]; a0 += m * w0[k]; a1 += m * w1[k]; }
            s_u.m.x[tid] = a0;
            s_u.m.z[tid] = a1;
        }
        __syncthreads();

        // ---- causal conv (circular buffer, per-channel, no cross-thread) ----
        {
            int pos = t & 3;
            s_conv[tid * 4 + pos] = s_u.m.x[tid];
            float xc = conv_b[tid];
            #pragma unroll
            for (int j2 = 0; j2 < DCONV; ++j2)
                xc += conv_w[tid * 4 + (3 - j2)] * s_conv[tid * 4 + ((pos - j2) & 3)];
            s_u.m.x[tid] = silu_f(xc);
        }
        __syncthreads();

        // ---- x_db = x @ x_proj_w.T  (288 outputs) ----
        if (tid < DTRANK + 2 * DSTATE) {
            const float* w = x_proj_w + (size_t)tid * DINNER;
            float acc = 0.0f;
            #pragma unroll 8
            for (int k = 0; k < DINNER; ++k) acc += s_u.m.x[k] * w[k];
            s_u.m.xdb[tid] = acc;
        }
        __syncthreads();

        // ---- dt (own channel), SSM state update + y ----
        {
            const float* w = dt_w + (size_t)tid * DTRANK;
            float acc = dt_b[tid];
            #pragma unroll
            for (int k = 0; k < DTRANK; ++k) acc += s_u.m.xdb[k] * w[k];
            float dtd = softplus_f(acc);
            float xd  = s_u.m.x[tid];
            const float* An = Aneg + (size_t)tid * DSTATE;
            float y = 0.0f;
            #pragma unroll 4
            for (int s2 = 0; s2 < DSTATE; ++s2) {
                float dA = __expf(dtd * An[s2]);
                float Bs = s_u.m.xdb[DTRANK + s2];
                float Cs = s_u.m.xdb[DTRANK + DSTATE + s2];
                float st = ssm[(size_t)s2 * DINNER + tid];
                st = st * dA + xd * dtd * Bs;
                ssm[(size_t)s2 * DINNER + tid] = st;
                y += st * Cs;
            }
            y += Dp[tid] * xd;
            float zd = s_u.m.z[tid];
            y *= silu_f(zd);
            __syncthreads();          // everyone done reading x/xdb before overwrite
            s_u.m.x[tid] = y;         // reuse x as y
        }
        __syncthreads();

        // ---- m_out + residual -> h; write fused[:,0:512] and carry prev ----
        if (tid < DMODEL) {
            const float* w = m_out_w + (size_t)tid * DINNER;
            float acc = 0.0f;
            #pragma unroll 8
            for (int k = 0; k < DINNER; ++k) acc += s_u.m.x[k] * w[k];
            float h = s_proj[tid] + acc;
            s_prev[tid] = h;
            fused_f16[((size_t)(b * SLEN + t)) * 1024 + tid] = (_Float16)h;
        }
        __syncthreads();
    }
}

// ---------------- host launch ----------------
extern "C" void kernel_launch(void* const* d_in, const int* in_sizes, int n_in,
                              void* d_out, int out_size, void* d_ws, size_t ws_size,
                              hipStream_t stream) {
    const float* tgt       = (const float*)d_in[0];
    const float* memory    = (const float*)d_in[1];
    // d_in[2..4]: spatial_shapes / level_start_index / valid_ratios (hard-coded constants)
    const float* enc       = (const float*)d_in[5];
    const float* fusion_w  = (const float*)d_in[6];
    const float* fusion_b  = (const float*)d_in[7];
    const float* n1g       = (const float*)d_in[8];
    const float* n1b       = (const float*)d_in[9];
    const float* n2g       = (const float*)d_in[10];
    const float* n2b       = (const float*)d_in[11];
    const float* q_w       = (const float*)d_in[12];
    const float* q_b       = (const float*)d_in[13];
    const float* kv_w      = (const float*)d_in[14];
    const float* kv_b      = (const float*)d_in[15];
    const float* cao_w     = (const float*)d_in[16];
    const float* cao_b     = (const float*)d_in[17];
    const float* outp_w    = (const float*)d_in[18];
    const float* outp_b    = (const float*)d_in[19];
    const float* in_proj_w = (const float*)d_in[20];
    const float* conv_w    = (const float*)d_in[21];
    const float* conv_b    = (const float*)d_in[22];
    const float* x_proj_w  = (const float*)d_in[23];
    const float* dt_w      = (const float*)d_in[24];
    const float* dt_b      = (const float*)d_in[25];
    const float* A_log     = (const float*)d_in[26];
    const float* Dp        = (const float*)d_in[27];
    const float* m_out_w   = (const float*)d_in[28];
    float* out = (float*)d_out;

    // ---- workspace carve-up (256B aligned) ----
    char* base = (char*)d_ws;
    size_t off = 0;
    auto carve = [&](size_t bytes) -> char* {
        char* p = base + off;
        off = (off + bytes + 255) & ~(size_t)255;
        return p;
    };
    const size_t nMem  = (size_t)ROWS_KV * DMODEL;           // 8,257,536
    const size_t nKvW  = (size_t)(2 * DMODEL) * DMODEL;      // 524,288
    const size_t nKV   = (size_t)BSZ * NHEADS * NKV * HDIM;  // 8,257,536
    const size_t nOutW = (size_t)DMODEL * DINNER;            // 524,288
    const size_t nFused = (size_t)BSZ * SLEN * 2 * DMODEL;   // 524,288

    _Float16* memF16   = (_Float16*)carve(nMem  * sizeof(_Float16));
    _Float16* kvwF16   = (_Float16*)carve(nKvW  * sizeof(_Float16));
    _Float16* Kf       = (_Float16*)carve(nKV   * sizeof(_Float16));
    _Float16* Vf       = (_Float16*)carve(nKV   * sizeof(_Float16));
    float*    Aneg     = (float*)   carve((size_t)DINNER * DSTATE * sizeof(float));
    float*    ssm      = (float*)   carve((size_t)BSZ * DSTATE * DINNER * sizeof(float));
    _Float16* fusedF16 = (_Float16*)carve(nFused * sizeof(_Float16));
    _Float16* outwF16  = (_Float16*)carve(nOutW * sizeof(_Float16));
    (void)ws_size; (void)in_sizes; (void)n_in; (void)out_size;

    // 1) casts + A_neg
    cast_f32_f16<<<(int)((nMem  + 255) / 256), 256, 0, stream>>>(memory, memF16, (int)nMem);
    cast_f32_f16<<<(int)((nKvW  + 255) / 256), 256, 0, stream>>>(kv_w,   kvwF16, (int)nKvW);
    cast_f32_f16<<<(int)((nOutW + 255) / 256), 256, 0, stream>>>(outp_w, outwF16, (int)nOutW);
    aneg_kernel<<<(DINNER * DSTATE + 255) / 256, 256, 0, stream>>>(A_log, Aneg, DINNER * DSTATE);

    // 2) KV projection GEMM (WMMA + TDM double-buffered LDS tiles)
    {
        int blocks = (ROWS_KV / BLK_M) * ((2 * DMODEL) / BLK_N); // 126 * 16 = 2016
        kv_gemm_wmma<<<blocks, 256, 0, stream>>>(memF16, kvwF16, kv_b, Kf, Vf);
    }

    // 3) sequential scan: one block per batch row
    scan_kernel<<<BSZ, 1024, 0, stream>>>(tgt, enc, fusion_w, fusion_b,
                                          n1g, n1b, n2g, n2b, q_w, q_b,
                                          cao_w, cao_b, in_proj_w, conv_w, conv_b,
                                          x_proj_w, dt_w, dt_b, Aneg, Dp, m_out_w,
                                          Kf, Vf, ssm, fusedF16);

    // 4) output projection GEMM (WMMA): 32 x 32 tiles, 8 waves/block
    {
        int totalWaves = 32 * 32;       // 1024
        int blocks = totalWaves / 8;    // 128
        out_gemm_wmma<<<blocks, 256, 0, stream>>>(fusedF16, outwF16, outp_b, out);
    }
}
